// SD_attn_withmoe_16131897164215
// MI455X (gfx1250) — matrix operand order repaired and checked by reading, not compiled, not run on hardware
//
#include <hip/hip_runtime.h>

// ---------------- config ----------------
#define USE_TDM 1          // Tensor Data Mover gather for MoE A-tile staging

// ---------------- constants ----------------
#define T_TOK   32768      // 8*64*64
#define CDIM    256
#define HIDR    128
#define NEXP    8
#define HEADS   8
#define HD      32
#define WSZ     8
#define NWIN    512        // 8 * 8 * 8
#define MAXTILE 2056       // T/16 + NEXP
#define ATT_SCALE 0.17677669529663687f   // 32^-0.5

// workspace layout (bytes)
#define XBF_OFF   0u
#define QKV_OFF   16777216u            // T*256*2
#define ATT_OFF   67108864u            // QKV_OFF + T*768*2
#define ROUTE_OFF 83886080u            // ATT_OFF + T*256*2
#define PMAX_OFF  84017152u
#define PERM_OFF  84148224u
#define META_OFF  84279296u            // 64 ints
#define WQKV_OFF  84280320u            // bf16 qkv_w: 8*768*256*2 = 3145728
#define WPRJ_OFF  87426048u            // bf16 proj_w: 8*256*256*2 = 1048576
// meta ints: [0..7]=counts [8..15]=cursor [16..24]=offsets [25..33]=tilePrefix [34]=totalTiles

typedef __attribute__((ext_vector_type(16))) __bf16 v16bf;
typedef __attribute__((ext_vector_type(8)))  float  v8f;
typedef __attribute__((ext_vector_type(4)))  unsigned int u32x4;
typedef __attribute__((ext_vector_type(8)))  int i32x8;
typedef __attribute__((ext_vector_type(4)))  int i32x4;

struct U8x4 { unsigned u[8]; };

static __device__ inline unsigned short f2bf(float f) {
  unsigned u = __builtin_bit_cast(unsigned, f);
  u += 0x7FFFu + ((u >> 16) & 1u);           // round-to-nearest-even
  return (unsigned short)(u >> 16);
}
static __device__ inline float bf2f(unsigned short h) {
  unsigned u = ((unsigned)h) << 16;
  return __builtin_bit_cast(float, u);
}
static __device__ inline unsigned pk(unsigned short lo, unsigned short hi) {
  return (unsigned)lo | ((unsigned)hi << 16);
}
static __device__ inline v16bf as_v16bf(U8x4 s) { return __builtin_bit_cast(v16bf, s); }
static __device__ inline v8f vzero8() {
  v8f z = {0.f,0.f,0.f,0.f,0.f,0.f,0.f,0.f};
  return z;
}
// A-fragment K pair base, contiguous K order (k = 16*khalf + s) matching B
static __device__ inline int a_kk(int v, int khalf) {
  return 16*khalf + 2*v;
}

// ---------------- kernels ----------------
__global__ void k_init_meta(int* meta) {
  if (threadIdx.x < 64) meta[threadIdx.x] = 0;
}

// one-shot f32 -> bf16 weight conversion (4 elems / thread)
__global__ void __launch_bounds__(256)
k_cvt_w(const float* __restrict__ src, unsigned short* __restrict__ dst, int n4) {
  int i = blockIdx.x * 256 + threadIdx.x;
  if (i >= n4) return;
  float4 f = *(const float4*)(src + (long)i*4);
  uint2 o;
  o.x = pk(f2bf(f.x), f2bf(f.y));
  o.y = pk(f2bf(f.z), f2bf(f.w));
  *(uint2*)(dst + (long)i*4) = o;
}

__global__ void __launch_bounds__(128)
k_router(const float* __restrict__ x,
         const float* __restrict__ w1, const float* __restrict__ b1,
         const float* __restrict__ w2, const float* __restrict__ b2,
         unsigned short* __restrict__ xbf, int* __restrict__ routes,
         float* __restrict__ pmax, int* __restrict__ meta) {
  __shared__ float xr[CDIM];
  __shared__ float hr[HIDR];
  __shared__ float lg[NEXP];
  const int t = blockIdx.x;
  const int tid = threadIdx.x;
  float v0 = x[(long)t*CDIM + tid];
  float v1 = x[(long)t*CDIM + 128 + tid];
  xr[tid] = v0; xr[tid + 128] = v1;
  xbf[(long)t*CDIM + tid]       = f2bf(v0);
  xbf[(long)t*CDIM + 128 + tid] = f2bf(v1);
  __syncthreads();
  float acc = 0.f;
  const float* wrow = w1 + tid*CDIM;
  #pragma unroll 4
  for (int k = 0; k < CDIM; ++k) acc += xr[k]*wrow[k];
  acc += b1[tid];
  hr[tid] = acc > 0.f ? acc : 0.f;
  __syncthreads();
  if (tid < NEXP) {
    float a = 0.f;
    const float* w2r = w2 + tid*HIDR;
    #pragma unroll 4
    for (int k = 0; k < HIDR; ++k) a += hr[k]*w2r[k];
    lg[tid] = a + b2[tid];
  }
  __syncthreads();
  if (tid == 0) {
    float m = lg[0]; int am = 0;
    for (int i = 1; i < NEXP; ++i) if (lg[i] > m) { m = lg[i]; am = i; }
    float s = 0.f;
    for (int i = 0; i < NEXP; ++i) s += __expf(lg[i] - m);
    routes[t] = am;
    pmax[t] = 1.0f / s;
    atomicAdd(&meta[am], 1);
  }
}

__global__ void k_prefix(int* meta) {
  if (threadIdx.x == 0) {
    int off = 0, tp = 0;
    for (int e = 0; e < NEXP; ++e) {
      meta[16 + e] = off; off += meta[e];
      meta[25 + e] = tp;  tp  += (meta[e] + 15) >> 4;
    }
    meta[24] = off;
    meta[33] = tp;
    meta[34] = tp;
  }
}

__global__ void k_scatter(const int* __restrict__ routes, int* meta, int* __restrict__ perm) {
  int t = blockIdx.x * 256 + threadIdx.x;
  int e = routes[t];
  int pos = meta[16 + e] + atomicAdd(&meta[8 + e], 1);
  perm[pos] = t;
}

// Grouped expert GEMM: Y[16 tokens x outDim] = Abf(tile) @ Wbf[e]^T + bias[e]
// CT = column tiles per wave (16 waves). MODE 0: bf16 out. MODE 1: f32 out * pmax.
template<int CT, int MODE>
__global__ void __launch_bounds__(512)
k_moe_gemm(const unsigned short* __restrict__ Abf,
           const unsigned short* __restrict__ Wbf, const float* __restrict__ bias,
           unsigned short* __restrict__ outBf, float* __restrict__ outF,
           const float* __restrict__ pmax,
           const int* __restrict__ meta, const int* __restrict__ perm,
           int outDim, int outLd) {
  __shared__ __align__(16) unsigned short aT[16*CDIM];
  __shared__ int stok[16];
  const int tileId = blockIdx.x;
  if (tileId >= meta[34]) return;
  int e = 0;
  while (e < NEXP-1 && tileId >= meta[25 + e + 1]) ++e;
  const int tIdx = tileId - meta[25 + e];
  const int base = meta[16 + e];
  const int cnt  = meta[e];
  const int tid  = threadIdx.x;
  const int wave = tid >> 5;
  if (tid < 16) {
    int slot = tIdx*16 + tid;
    stok[tid] = (slot < cnt) ? perm[base + slot] : -1;
  }
  __syncthreads();
#if USE_TDM
  if (wave == 0) {
    // ---- Tensor Data Mover gather: 16 token rows (512B each) -> LDS ----
    int vc = cnt - tIdx*16; if (vc > 16) vc = 16;      // valid rows (first vc slots)
    unsigned ldsa = (unsigned)(unsigned long long)(size_t)aT;  // flat LDS addr truncates to LDS offset
    unsigned long long ga = (unsigned long long)(size_t)Abf;
    u32x4 g0;
    g0.x = 0x80000001u;                                 // count=1, gather_mode=1, 16-bit idx
    g0.y = ldsa;                                        // lds_addr
    g0.z = (unsigned)ga;                                // global_addr[31:0]
    g0.w = (unsigned)((ga >> 32) & 0x01FFFFFFu) | (2u << 30);  // global_addr[56:32], type=2
    i32x8 g1;
    g1.s0 = (int)(1u << 16);                            // wg_mask=0, data_size=1 (2B)
    g1.s1 = (int)(256u << 16);                          // tensor_dim0[15:0]=256 @ bits63:48
    g1.s2 = (int)(0x8000u << 16);                       // dim0 hi=0; tensor_dim1 lo16=32768
    g1.s3 = (int)(256u << 16);                          // dim1 hi=0; tile_dim0=256
    g1.s4 = vc;                                         // tile_dim1 = #valid indices; tile_dim2=0
    g1.s5 = 256;                                        // tensor_dim0_stride lo32 = 256
    g1.s6 = 0; g1.s7 = 0;
    i32x4 g2, g3;
    int tw[8];
    #pragma unroll
    for (int i = 0; i < 8; ++i) {
      int t0 = stok[2*i],  t1 = stok[2*i + 1];
      if (t0 < 0) t0 = 0;
      if (t1 < 0) t1 = 0;
      tw[i] = (int)pk((unsigned short)t0, (unsigned short)t1);
    }
    g2.x = tw[0]; g2.y = tw[1]; g2.z = tw[2]; g2.w = tw[3];
    g3.x = tw[4]; g3.y = tw[5]; g3.z = tw[6]; g3.w = tw[7];
    i32x8 g4;                                            // extra group (zero-filled)
    g4.s0 = 0; g4.s1 = 0; g4.s2 = 0; g4.s3 = 0;
    g4.s4 = 0; g4.s5 = 0; g4.s6 = 0; g4.s7 = 0;
    __builtin_amdgcn_tensor_load_to_lds(g0, g1, g2, g3, g4, 0);
    __builtin_amdgcn_s_wait_tensorcnt(0);
  }
  __syncthreads();
#else
  { // manual stage A tile (16 x 256 bf16) to LDS: 512 threads x 16B
    int r = tid >> 5, ch = tid & 31;
    int tok = stok[r];
    uint4 val = make_uint4(0u,0u,0u,0u);
    if (tok >= 0) val = *(const uint4*)(Abf + (long)tok*CDIM + ch*8);
    *((uint4*)(aT + r*CDIM + ch*8)) = val;
  }
  __syncthreads();
#endif
  const int lane  = tid & 31;
  const int nl    = lane & 15;
  const int khalf = lane >> 4;
  v8f acc[CT];
  #pragma unroll
  for (int c = 0; c < CT; ++c) acc[c] = vzero8();
  const unsigned short* We = Wbf + (long)e * outDim * CDIM;
  #pragma unroll
  for (int ks = 0; ks < CDIM/32; ++ks) {
    const int kbase = ks*32;
    U8x4 as;
    #pragma unroll
    for (int v = 0; v < 8; ++v)
      as.u[v] = *(const unsigned*)(aT + nl*CDIM + kbase + a_kk(v, khalf));
    v16bf afrag = as_v16bf(as);
    #pragma unroll
    for (int c = 0; c < CT; ++c) {
      int colGlob = (wave*CT + c)*16 + nl;
      const unsigned short* wr = We + (long)colGlob*CDIM + kbase + khalf*16;
      uint4 p0 = *(const uint4*)(wr);
      uint4 p1 = *(const uint4*)(wr + 8);
      U8x4 bs;
      bs.u[0]=p0.x; bs.u[1]=p0.y; bs.u[2]=p0.z; bs.u[3]=p0.w;
      bs.u[4]=p1.x; bs.u[5]=p1.y; bs.u[6]=p1.z; bs.u[7]=p1.w;
      acc[c] = __builtin_amdgcn_wmma_f32_16x16x32_bf16(
          false, afrag, false, as_v16bf(bs), (short)0, acc[c], false, false);
    }
  }
  #pragma unroll
  for (int c = 0; c < CT; ++c) {
    int colGlob = (wave*CT + c)*16 + nl;
    float bv = bias[e*outDim + colGlob];
    #pragma unroll
    for (int r = 0; r < 8; ++r) {
      int M = r + 8*khalf;
      int tok = stok[M];
      if (tok < 0) continue;
      float val = acc[c][r] + bv;
      if (MODE == 0) outBf[(long)tok*outLd + colGlob] = f2bf(val);
      else           outF [(long)tok*outLd + colGlob] = val * pmax[tok];
    }
  }
}

// One block per (window, head). qkv bf16 layout [t][ q(256) | k(256) | v(256) ].
__global__ void __launch_bounds__(256)
k_attn(const unsigned short* __restrict__ qkv, unsigned short* __restrict__ attnOut) {
  __shared__ __align__(16) unsigned short qS[64*HD], kS[64*HD], vS[64*HD];
  __shared__ float Sm[64*64];
  const int wid  = blockIdx.x >> 3;
  const int head = blockIdx.x & 7;
  const int b  = wid >> 6;
  const int wy = (wid >> 3) & 7;
  const int wx = wid & 7;
  const int tid = threadIdx.x;
  { // load + RoPE: thread -> token m, channel quarter (8 chans = 4 pairs)
    const int m = tid >> 2;
    const int quarter = tid & 3;
    const int yy = m >> 3, xx = m & 7;
    const int t = ((b*64 + wy*WSZ + yy) << 6) + wx*WSZ + xx;
    const long qb = (long)t*768 + head*HD;
    const int ch0 = quarter*8;
    *((uint4*)(vS + m*HD + ch0)) = *(const uint4*)(qkv + qb + 512 + ch0);
    #pragma unroll
    for (int pp = 0; pp < 4; ++pp) {
      int P  = (ch0 >> 1) + pp;          // pair 0..15
      int c0 = 2*P;                      // channel (same formula both halves)
      int p  = P & 7;
      float inv = __powf(10000.f, -((float)(2*p)) / 16.f);
      float ang = ((P < 8) ? (float)yy : (float)xx) * inv;
      float sn, cs;
      __sincosf(ang, &sn, &cs);
      float q0 = bf2f(qkv[qb + c0]),       q1 = bf2f(qkv[qb + c0 + 1]);
      float k0 = bf2f(qkv[qb + 256 + c0]), k1 = bf2f(qkv[qb + 256 + c0 + 1]);
      qS[m*HD + c0]     = f2bf((q0*cs - q1*sn) * ATT_SCALE);
      qS[m*HD + c0 + 1] = f2bf((q0*sn + q1*cs) * ATT_SCALE);
      kS[m*HD + c0]     = f2bf(k0*cs - k1*sn);
      kS[m*HD + c0 + 1] = f2bf(k0*sn + k1*cs);
    }
  }
  __syncthreads();
  const int wave = tid >> 5, lane = tid & 31;
  const int nl = lane & 15, khalf = lane >> 4;
  // S = (q*scale) @ k^T : 4x4 tiles of 16x16, K=32 (one WMMA each)
  #pragma unroll
  for (int tt = 0; tt < 2; ++tt) {
    int ti = wave + tt*8;
    int mi = ti >> 2, ni = ti & 3;
    int m = mi*16 + nl;
    U8x4 as;
    #pragma unroll
    for (int v = 0; v < 8; ++v)
      as.u[v] = *(const unsigned*)(qS + m*HD + a_kk(v, khalf));
    int n = ni*16 + nl;
    U8x4 bs;
    uint4 bb0 = *(const uint4*)(kS + n*HD + khalf*16);
    uint4 bb1 = *(const uint4*)(kS + n*HD + khalf*16 + 8);
    bs.u[0]=bb0.x; bs.u[1]=bb0.y; bs.u[2]=bb0.z; bs.u[3]=bb0.w;
    bs.u[4]=bb1.x; bs.u[5]=bb1.y; bs.u[6]=bb1.z; bs.u[7]=bb1.w;
    v8f accS = vzero8();
    accS = __builtin_amdgcn_wmma_f32_16x16x32_bf16(
        false, as_v16bf(as), false, as_v16bf(bs), (short)0, accS, false, false);
    #pragma unroll
    for (int r = 0; r < 8; ++r)
      Sm[(mi*16 + r + 8*khalf)*64 + ni*16 + nl] = accS[r];
  }
  __syncthreads();
  if (tid < 64) { // row softmax
    float* row = Sm + tid*64;
    float m = row[0];
    for (int i = 1; i < 64; ++i) m = fmaxf(m, row[i]);
    float s = 0.f;
    for (int i = 0; i < 64; ++i) { float e2 = __expf(row[i]-m); row[i] = e2; s += e2; }
    float inv = 1.f / s;
    for (int i = 0; i < 64; ++i) row[i] *= inv;
  }
  __syncthreads();
  { // O = P @ v : 4x2 tiles, one per wave, K=64 -> 2 WMMA steps
    int mi = wave >> 1, ni = wave & 1;
    v8f acc = vzero8();
    #pragma unroll
    for (int ks = 0; ks < 2; ++ks) {
      int kbase = ks*32;
      int m = mi*16 + nl;
      U8x4 as;
      #pragma unroll
      for (int v = 0; v < 8; ++v) {
        int kk = kbase + a_kk(v, khalf);
        as.u[v] = pk(f2bf(Sm[m*64 + kk]), f2bf(Sm[m*64 + kk + 1]));
      }
      int n = ni*16 + nl;
      U8x4 bs;
      #pragma unroll
      for (int i = 0; i < 16; i += 2) {
        unsigned short e0 = vS[(kbase + khalf*16 + i)     * HD + n];
        unsigned short e1 = vS[(kbase + khalf*16 + i + 1) * HD + n];
        bs.u[i >> 1] = pk(e0, e1);
      }
      acc = __builtin_amdgcn_wmma_f32_16x16x32_bf16(
          false, as_v16bf(as), false, as_v16bf(bs), (short)0, acc, false, false);
    }
    #pragma unroll
    for (int r = 0; r < 8; ++r) {
      int m  = mi*16 + r + 8*khalf;
      int yy = m >> 3, xx = m & 7;
      int t  = ((b*64 + wy*WSZ + yy) << 6) + wx*WSZ + xx;
      int cc = head*HD + ni*16 + nl;
      attnOut[(long)t*CDIM + cc] = f2bf(acc[r]);
    }
  }
}

// ---------------- launcher ----------------
extern "C" void kernel_launch(void* const* d_in, const int* in_sizes, int n_in,
                              void* d_out, int out_size, void* d_ws, size_t ws_size,
                              hipStream_t stream) {
  (void)in_sizes; (void)n_in; (void)out_size; (void)ws_size;
  const float* x         = (const float*)d_in[0];
  const float* router_w1 = (const float*)d_in[1];
  const float* router_b1 = (const float*)d_in[2];
  const float* router_w2 = (const float*)d_in[3];
  const float* router_b2 = (const float*)d_in[4];
  const float* qkv_w     = (const float*)d_in[5];
  const float* qkv_b     = (const float*)d_in[6];
  const float* proj_w    = (const float*)d_in[7];
  const float* proj_b    = (const float*)d_in[8];

  char* ws = (char*)d_ws;
  unsigned short* xbf    = (unsigned short*)(ws + XBF_OFF);
  unsigned short* qkvbf  = (unsigned short*)(ws + QKV_OFF);
  unsigned short* attnbf = (unsigned short*)(ws + ATT_OFF);
  int*   routes = (int*)  (ws + ROUTE_OFF);
  float* pmax   = (float*)(ws + PMAX_OFF);
  int*   perm   = (int*)  (ws + PERM_OFF);
  int*   meta   = (int*)  (ws + META_OFF);
  unsigned short* wqkvbf = (unsigned short*)(ws + WQKV_OFF);
  unsigned short* wprjbf = (unsigned short*)(ws + WPRJ_OFF);

  k_init_meta<<<1, 64, 0, stream>>>(meta);
  { // weight conversion: n/4 threads
    int n4q = NEXP*768*CDIM/4, n4p = NEXP*CDIM*CDIM/4;
    k_cvt_w<<<(n4q + 255)/256, 256, 0, stream>>>(qkv_w,  wqkvbf, n4q);
    k_cvt_w<<<(n4p + 255)/256, 256, 0, stream>>>(proj_w, wprjbf, n4p);
  }
  k_router<<<T_TOK, 128, 0, stream>>>(x, router_w1, router_b1, router_w2, router_b2,
                                      xbf, routes, pmax, meta);
  k_prefix<<<1, 32, 0, stream>>>(meta);
  k_scatter<<<T_TOK/256, 256, 0, stream>>>(routes, meta, perm);
  k_moe_gemm<3,0><<<MAXTILE, 512, 0, stream>>>(xbf, wqkvbf, qkv_b,
                                               qkvbf, nullptr, nullptr,
                                               meta, perm, 768, 768);
  k_attn<<<NWIN*HEADS, 256, 0, stream>>>(qkvbf, attnbf);
  k_moe_gemm<1,1><<<MAXTILE, 512, 0, stream>>>(attnbf, wprjbf, proj_b,
                                               nullptr, (float*)d_out, pmax,
                                               meta, perm, 256, 256);
}
